// GModel_19636590478071
// MI455X (gfx1250) — matrix operand order repaired and checked by reading, not compiled, not run on hardware
//
#include <hip/hip_runtime.h>

// ---------------- problem constants (from reference) ----------------
#define GORDER   8
#define IN_DIM   32768
#define CLASSES  10
#define BATCH    2048
// THETA = 2*pi/8 ; hardware v_cos/v_sin take revolutions => rev = v * (1/8)
#define REV_SCALE 0.125f
#define OUT_SCALE 0.0055242717280199026f   // 1/sqrt(32768)

// ---------------- tiling ----------------
#define KTILES      (IN_DIM / 32)   // 1024 WMMA K-tiles of 32
#define KP          16              // K-partitions (grid.y)
#define WAVES       8               // waves per block (256 threads, wave32)
#define TILES_PER_BLOCK (KTILES / KP)           // 64
#define TILES_PER_WAVE  (TILES_PER_BLOCK / WAVES) // 8
#define MTILES      (BATCH / 16)    // 128

typedef __attribute__((ext_vector_type(16))) _Float16 v16h;
typedef __attribute__((ext_vector_type(8)))  float    v8f;
typedef __attribute__((ext_vector_type(4)))  float    f4;

// --------------------------------------------------------------------
// Kernel 1: build WMMA B-fragments for cos(theta*round(w)) / sin(...)
// Layout per K-tile t, per lane l (wave32 B-matrix 32x16, 16-bit):
//   n = l & 15 (class, padded to 16), K = t*32 + (j&7) + (j>=8?16:0) + ((l<16) ? 0 : 8)
// Stored as [t][lane][0..15]=cos, [t][lane][16..31]=sin  (f16)
// --------------------------------------------------------------------
__global__ void build_b(const float* __restrict__ weight,
                        _Float16* __restrict__ bpack) {
    int idx = blockIdx.x * blockDim.x + threadIdx.x;   // one thread per (t,lane)
    if (idx >= KTILES * 32) return;
    int t = idx >> 5;
    int l = idx & 31;
    int n = l & 15;
    int kbase = t * 32 + ((l < 16) ? 0 : 8);

    v16h cv = {};
    v16h sv = {};
    if (n < CLASSES) {
        const float* wrow = weight + (size_t)n * IN_DIM + kbase;
#pragma unroll
        for (int j = 0; j < 16; ++j) {
            float wq  = __builtin_rintf(wrow[j + ((j >= 8) ? 8 : 0)]);      // round-half-even (matches jnp.round)
            float rev = wq * REV_SCALE;
            cv[j] = (_Float16)__builtin_amdgcn_cosf(rev);
            sv[j] = (_Float16)__builtin_amdgcn_sinf(rev);
        }
    }
    _Float16* dst = bpack + (size_t)idx * 32;
    *(v16h*)(dst)      = cv;
    *(v16h*)(dst + 16) = sv;
}

// --------------------------------------------------------------------
// Kernel 2: main fused trig + WMMA GEMM.
//   partial[kp][row][n16] = sum over K-chunk of cos(tx)cos(tw)+sin(tx)sin(tw)
// A-fragment (16-bit A 16x32): lane m = l&15, K = (j&7) + (j>=8?16:0) + (l<16?0:8)
// x is streamed once (256 MB) -> non-temporal loads, keep L2 for bpack reuse.
// --------------------------------------------------------------------
__launch_bounds__(256)
__global__ void groupsim_main(const float* __restrict__ x,
                              const _Float16* __restrict__ bpack,
                              float* __restrict__ partial) {
    __shared__ float smem[WAVES][32][8];

    const int mtile = blockIdx.x;       // 0..127
    const int kp    = blockIdx.y;       // 0..15
    const int tid   = threadIdx.x;
    const int wave  = tid >> 5;
    const int lane  = tid & 31;
    const int m     = lane & 15;
    const int koff  = (lane < 16) ? 0 : 8;

    const float* rowPtr = x + (size_t)(mtile * 16 + m) * IN_DIM;

    v8f acc = {};
    const int tbase = kp * TILES_PER_BLOCK;

#pragma unroll 2
    for (int i = 0; i < TILES_PER_WAVE; ++i) {
        const int t  = tbase + wave + i * WAVES;   // strided across waves
        const int k0 = t * 32;

        // A fragment: two contiguous 8-float runs per lane (4x b128 NT loads)
        const f4* p0 = (const f4*)(rowPtr + k0 + koff);
        const f4* p1 = (const f4*)(rowPtr + k0 + 16 + koff);
        f4 a0 = __builtin_nontemporal_load(p0);
        f4 a1 = __builtin_nontemporal_load(p0 + 1);
        f4 a2 = __builtin_nontemporal_load(p1);
        f4 a3 = __builtin_nontemporal_load(p1 + 1);

        float v[16];
#pragma unroll
        for (int j = 0; j < 4; ++j) { v[j] = a0[j]; v[4 + j] = a1[j]; v[8 + j] = a2[j]; v[12 + j] = a3[j]; }

        v16h ca, sa;
#pragma unroll
        for (int j = 0; j < 16; ++j) {
            float rev = v[j] * REV_SCALE;
            ca[j] = (_Float16)__builtin_amdgcn_cosf(rev);
            sa[j] = (_Float16)__builtin_amdgcn_sinf(rev);
        }

        // pre-swizzled B fragments: 64 contiguous bytes per lane (RT -> L2 reuse)
        const v16h* bp = (const v16h*)(bpack + ((size_t)t * 32 + lane) * 32);
        v16h cb = bp[0];
        v16h sb = bp[1];

        acc = __builtin_amdgcn_wmma_f32_16x16x32_f16(
            false, ca, false, cb, (short)0, acc, false, false);
        acc = __builtin_amdgcn_wmma_f32_16x16x32_f16(
            false, sa, false, sb, (short)0, acc, false, false);
    }

    // cross-wave reduction through LDS (fixed order -> deterministic)
#pragma unroll
    for (int r = 0; r < 8; ++r) smem[wave][lane][r] = acc[r];
    __syncthreads();

    // C/D layout: lane l, reg r -> M = r + 8*(l>=16), N = l&15
    const int om = tid >> 4;            // 0..15 (tile row)
    const int on = tid & 15;            // 0..15 (class, padded)
    const int sl = on + ((om >= 8) ? 16 : 0);
    const int sr = om & 7;
    float sum = 0.0f;
#pragma unroll
    for (int w = 0; w < WAVES; ++w) sum += smem[w][sl][sr];

    // partial tile is write-once/read-once: keep it out of near caches
    __builtin_nontemporal_store(
        sum, &partial[((size_t)kp * BATCH + mtile * 16 + om) * 16 + on]);
}

// --------------------------------------------------------------------
// Kernel 3: fold K-partitions in fixed order, apply radius * OUT_SCALE
// --------------------------------------------------------------------
__global__ void reduce_out(const float* __restrict__ partial,
                           const float* __restrict__ radius,
                           float* __restrict__ out) {
    int idx = blockIdx.x * blockDim.x + threadIdx.x;
    if (idx >= BATCH * CLASSES) return;
    int row = idx / CLASSES;
    int n   = idx % CLASSES;
    float s = 0.0f;
#pragma unroll
    for (int kp = 0; kp < KP; ++kp)
        s += __builtin_nontemporal_load(&partial[((size_t)kp * BATCH + row) * 16 + n]);
    out[idx] = s * radius[0] * OUT_SCALE;
}

// --------------------------------------------------------------------
extern "C" void kernel_launch(void* const* d_in, const int* in_sizes, int n_in,
                              void* d_out, int out_size, void* d_ws, size_t ws_size,
                              hipStream_t stream) {
    const float* x      = (const float*)d_in[0];   // [2048, 32768] f32
    const float* weight = (const float*)d_in[1];   // [10, 32768] f32
    const float* radius = (const float*)d_in[2];   // [1] f32
    float* out = (float*)d_out;                    // [2048, 10] f32

    // workspace: B fragments (2 MB) then partials (2 MB)
    _Float16* bpack   = (_Float16*)d_ws;
    float*    partial = (float*)((char*)d_ws + (size_t)KTILES * 32 * 32 * sizeof(_Float16));

    build_b<<<(KTILES * 32 + 255) / 256, 256, 0, stream>>>(weight, bpack);

    dim3 grid(MTILES, KP);
    groupsim_main<<<grid, 256, 0, stream>>>(x, bpack, partial);

    reduce_out<<<(BATCH * CLASSES + 255) / 256, 256, 0, stream>>>(partial, radius, out);
}
